// MixOfExperts_80453327389116
// MI455X (gfx1250) — compile-verified, matrix operand order checked by reading, not run on hardware
//
#include <hip/hip_runtime.h>
#include <hip/hip_bf16.h>

// MoE fused kernel for gfx1250 (MI455X), wave32 + V_WMMA_F32_16X16X32_BF16.
//
// B=32,T=512 -> 16384 tokens; F=128 (K); U=64; E=32 experts.
// out[t,u] = sum_e leaky( x[t,:]@We[e][:,u] + be[e,u] ) * ( x[t,:]@Wg[:,e] + bg[e] )
//
// Two-phase plan:
//   prep kernel: convert x/Wg/We fp32 -> bf16 into d_ws, already laid out in
//                per-lane WMMA fragment order (zero conversion/transpose work
//                in the hot kernel).
//   main kernel: 128 threads (4 wave32) per WG -> 256 WGs for WGP spread.
//                Per wave: 16-token tile, A fragments register-resident for
//                all K; per-expert B fragments double-buffered in LDS via
//                pure b128 copies; 16 WMMAs per expert per wave.

typedef __attribute__((ext_vector_type(16))) __bf16          v16bf;
typedef __attribute__((ext_vector_type(16))) unsigned short  v16u16;
typedef __attribute__((ext_vector_type(8)))  unsigned short  v8u16;
typedef __attribute__((ext_vector_type(8)))  float           v8f;

#define F_DIM 128
#define U_DIM 64
#define E_DIM 32
#define TOKENS (32 * 512)

// ws layout (bf16 element offsets)
#define XF_ELEMS  (TOKENS * F_DIM)            // 2,097,152  x fragments [tile][kt][lane][16]
#define WEF_ELEMS (E_DIM * 4 * 4 * 32 * 16)   //   262,144  We fragments [e][kt][nt][lane][16]
#define WGF_ELEMS (4 * 2 * 32 * 16)           //     4,096  Wg fragments [kt][g][lane][16]
#define WS_ELEMS  (XF_ELEMS + WEF_ELEMS + WGF_ELEMS)
#define WS_BYTES  ((size_t)WS_ELEMS * 2)

// fp32 -> bf16 round-to-nearest-even (bit trick; inputs are finite normals)
__device__ __forceinline__ unsigned short f2bf(float f) {
  unsigned int u = __float_as_uint(f);
  return (unsigned short)((u + 0x7FFFu + ((u >> 16) & 1u)) >> 16);
}

__device__ __forceinline__ v16bf bfcast(v16u16 u) {
  union { v16u16 u; v16bf b; } cv; cv.u = u; return cv.b;
}

__device__ __forceinline__ v8f wmma_bf16(v16bf a, v16bf b, v8f c) {
  return __builtin_amdgcn_wmma_f32_16x16x32_bf16(
      /*neg_a=*/false, a, /*neg_b=*/false, b,
      /*c_mod=*/(short)0, c, /*reuse_a=*/false, /*reuse_b=*/false);
}

// Fragment K index for element i within a lane's v16bf:
//   elements 0..7  -> K = kbase + h*8 + i
//   elements 8..15 -> K = kbase + 16 + h*8 + (i-8)
__device__ __forceinline__ int frag_k(int kt, int h, int i) {
  return kt * 32 + h * 8 + (i & 7) + ((i & 8) ? 16 : 0);
}

// ---------------------------------------------------------------------------
// Prep: build bf16 fragment-ordered copies of x, We, Wg in workspace.
// ---------------------------------------------------------------------------
__global__ __launch_bounds__(256) void moe_prep_kernel(
    const float* __restrict__ x, const float* __restrict__ Wg,
    const float* __restrict__ We, unsigned short* __restrict__ ws) {
  int idx = blockIdx.x * 256 + threadIdx.x;
  if (idx < XF_ELEMS) {
    // xf[tile][kt][lane][16] : A fragment, M = lane&15 within 16-token tile
    int i    = idx & 15;
    int l    = (idx >> 4) & 31;
    int kt   = (idx >> 9) & 3;
    int tile = idx >> 11;
    int n = l & 15, h = l >> 4;
    int token = tile * 16 + n;
    int K = frag_k(kt, h, i);
    ws[idx] = f2bf(x[token * F_DIM + K]);
  } else if (idx < XF_ELEMS + WEF_ELEMS) {
    // wef[e][kt][nt][lane][16] : B fragment, N(=u) = nt*16 + (lane&15)
    int r  = idx - XF_ELEMS;
    int i  = r & 15;
    int l  = (r >> 4) & 31;
    int nt = (r >> 9) & 3;
    int kt = (r >> 11) & 3;
    int e  = r >> 13;
    int n = l & 15, h = l >> 4;
    int u = nt * 16 + n;
    int K = frag_k(kt, h, i);
    ws[idx] = f2bf(We[e * (F_DIM * U_DIM) + K * U_DIM + u]);
  } else if (idx < WS_ELEMS) {
    // wgf[kt][g][lane][16] : B fragment, N(=expert) = g*16 + (lane&15)
    int r  = idx - XF_ELEMS - WEF_ELEMS;
    int i  = r & 15;
    int l  = (r >> 4) & 31;
    int g  = (r >> 9) & 1;
    int kt = r >> 10;
    int n = l & 15, h = l >> 4;
    int ecol = g * 16 + n;
    int K = frag_k(kt, h, i);
    ws[idx] = f2bf(Wg[K * E_DIM + ecol]);
  }
}

// ---------------------------------------------------------------------------
// Main: fragment loads only; double-buffered LDS staging of We per expert.
// 128 threads = 4 wave32; 256 workgroups for WGP spread (1024 waves total).
// ---------------------------------------------------------------------------
__global__ __launch_bounds__(128) void moe_main_kernel(
    const unsigned short* __restrict__ ws,
    const float* __restrict__ bg,   // [32]
    const float* __restrict__ be,   // [32, 64]
    float* __restrict__ out)        // [16384, 64]
{
  // 2 buffers x 8192 bf16 (16 KB) for one expert's 16 B-fragments
  __shared__ __align__(16) unsigned short ldsWe[2][4 * 4 * 32 * 16];

  const unsigned short* xf  = ws;
  const unsigned short* wef = ws + XF_ELEMS;
  const unsigned short* wgf = ws + XF_ELEMS + WEF_ELEMS;

  const int tid  = threadIdx.x;
  const int lane = tid & 31;
  const int wave = tid >> 5;
  const int h    = lane >> 4;
  const int n    = lane & 15;

  const int tile = blockIdx.x * 4 + wave;   // 16-token tile per wave
  const int tw   = tile * 16;

  // ---- A fragments (held for all K) : 2 coalesced b128 loads each ----
  v16bf a[4];
#pragma unroll
  for (int kt = 0; kt < 4; ++kt)
    a[kt] = bfcast(*(const v16u16*)(xf + ((size_t)(tile * 4 + kt) * 32 + lane) * 16));

  // ---- Gate tile [16 tok x 32 experts] ----
  v8f g0 = {}, g1 = {};
#pragma unroll
  for (int kt = 0; kt < 4; ++kt) {
    v16bf b0 = bfcast(*(const v16u16*)(wgf + ((kt * 2 + 0) * 32 + lane) * 16));
    v16bf b1 = bfcast(*(const v16u16*)(wgf + ((kt * 2 + 1) * 32 + lane) * 16));
    g0 = wmma_bf16(a[kt], b0, g0);
    g1 = wmma_bf16(a[kt], b1, g1);
  }
  {
    float b0 = bg[n], b1 = bg[16 + n];
#pragma unroll
    for (int j = 0; j < 8; ++j) { g0[j] += b0; g1[j] += b1; }
  }

  v8f zero = {};
  v8f oacc[4];
#pragma unroll
  for (int nt = 0; nt < 4; ++nt) oacc[nt] = zero;

  // ---- Preload expert 0 into LDS buffer 0 (pure b128 memcpy) ----
  {
    const uint4* src = (const uint4*)wef;            // 1024 uint4 per expert
    uint4* dst = (uint4*)&ldsWe[0][0];
#pragma unroll
    for (int c = 0; c < 8; ++c) dst[tid + c * 128] = src[tid + c * 128];
  }
  __syncthreads();

  int buf = 0;
  for (int e = 0; e < E_DIM; ++e) {
    // Issue global loads for next expert early (latency hidden by WMMAs)
    uint4 nxt[8];
    if (e + 1 < E_DIM) {
      const uint4* src = (const uint4*)wef + (size_t)(e + 1) * 1024;
#pragma unroll
      for (int c = 0; c < 8; ++c) nxt[c] = src[tid + c * 128];
    }

    // h tile [16 tok x 64 U]: 16 WMMAs, B from LDS (lane*32B, conflict-free)
    const unsigned short* lb = &ldsWe[buf][0];
    v8f hacc[4];
#pragma unroll
    for (int nt = 0; nt < 4; ++nt) hacc[nt] = zero;
#pragma unroll
    for (int kt = 0; kt < 4; ++kt) {
#pragma unroll
      for (int nt = 0; nt < 4; ++nt)
        hacc[nt] = wmma_bf16(
            a[kt],
            bfcast(*(const v16u16*)(lb + ((kt * 4 + nt) * 32 + lane) * 16)),
            hacc[nt]);
    }

    // Store next expert into the other buffer (previous expert's buffer is
    // free: the barrier at the end of the prior iteration retired all reads).
    if (e + 1 < E_DIM) {
      uint4* dst = (uint4*)&ldsWe[buf ^ 1][0];
#pragma unroll
      for (int c = 0; c < 8; ++c) dst[tid + c * 128] = nxt[c];
    }

    // gate[m,e] broadcast matching C/D layout: src lane (same half) holds N=e&15
    float gl[8];
#pragma unroll
    for (int j = 0; j < 8; ++j) {
      float gv = (e < 16) ? g0[j] : g1[j];            // uniform select
      gl[j] = __shfl(gv, (lane & 16) | (e & 15), 32);
    }

    // bias + LeakyReLU + weighted accumulate
#pragma unroll
    for (int nt = 0; nt < 4; ++nt) {
      float bev = be[e * U_DIM + nt * 16 + n];
#pragma unroll
      for (int j = 0; j < 8; ++j) {
        float hv = hacc[nt][j] + bev;
        hv = (hv >= 0.0f) ? hv : 0.01f * hv;
        oacc[nt][j] = __builtin_fmaf(hv, gl[j], oacc[nt][j]);
      }
    }

    __syncthreads();   // next expert staged + all reads of `buf` retired
    buf ^= 1;
  }

  // ---- Store out: token = tw + j + 8*h, u = nt*16 + n ----
  float* op = out + (size_t)tw * U_DIM;
#pragma unroll
  for (int nt = 0; nt < 4; ++nt)
#pragma unroll
    for (int j = 0; j < 8; ++j)
      op[(j + 8 * h) * U_DIM + nt * 16 + n] = oacc[nt][j];
}

// ---------------------------------------------------------------------------
// Fallback (no workspace): fused kernel, conversions in-kernel.
// ---------------------------------------------------------------------------
#define LDS_STRIDE 136

__device__ __forceinline__ v16bf load_a_f32(const float* __restrict__ p0,
                                            const float* __restrict__ p1) {
  v16u16 u;
#pragma unroll
  for (int i = 0; i < 8; ++i) { u[i] = f2bf(p0[i]); u[i + 8] = f2bf(p1[i]); }
  return bfcast(u);
}

__device__ __forceinline__ v16bf load_b_lds(const unsigned short* row, int kt, int h) {
  v8u16 lo = *(const v8u16*)(row + kt * 32 + h * 8);
  v8u16 hi = *(const v8u16*)(row + kt * 32 + 16 + h * 8);
  v16u16 u = __builtin_shufflevector(lo, hi, 0,1,2,3,4,5,6,7,8,9,10,11,12,13,14,15);
  return bfcast(u);
}

__global__ __launch_bounds__(256) void moe_fallback_kernel(
    const float* __restrict__ x, const float* __restrict__ Wg,
    const float* __restrict__ bg, const float* __restrict__ We,
    const float* __restrict__ be, float* __restrict__ out) {
  __shared__ __align__(16) unsigned short ldsWg[E_DIM * LDS_STRIDE];
  __shared__ __align__(16) unsigned short ldsWe[U_DIM * LDS_STRIDE];

  const int tid = threadIdx.x, lane = tid & 31, wave = tid >> 5;
  const int h = lane >> 4, n = lane & 15;

#pragma unroll
  for (int i = 0; i < 16; ++i) {
    int idx = tid + i * 256;
    int f = idx >> 5, e = idx & 31;
    ldsWg[e * LDS_STRIDE + f] = f2bf(Wg[f * E_DIM + e]);
  }
  __syncthreads();

  const int tw = blockIdx.x * 128 + wave * 16;
  const float* xrow = x + (size_t)(tw + n) * F_DIM;
  v16bf a[4];
#pragma unroll
  for (int kt = 0; kt < 4; ++kt)
    a[kt] = load_a_f32(xrow + kt * 32 + h * 8, xrow + kt * 32 + 16 + h * 8);

  v8f g0 = {}, g1 = {};
#pragma unroll
  for (int kt = 0; kt < 4; ++kt) {
    g0 = wmma_bf16(a[kt], load_b_lds(ldsWg + n * LDS_STRIDE, kt, h), g0);
    g1 = wmma_bf16(a[kt], load_b_lds(ldsWg + (16 + n) * LDS_STRIDE, kt, h), g1);
  }
  { float b0 = bg[n], b1 = bg[16 + n];
#pragma unroll
    for (int j = 0; j < 8; ++j) { g0[j] += b0; g1[j] += b1; } }

  v8f zero = {};
  v8f oacc[4];
#pragma unroll
  for (int nt = 0; nt < 4; ++nt) oacc[nt] = zero;

  for (int e = 0; e < E_DIM; ++e) {
    __syncthreads();
    const float* wep = We + (size_t)e * (F_DIM * U_DIM);
#pragma unroll
    for (int i = 0; i < 32; ++i) {
      int idx = tid + i * 256;
      int f = idx >> 6, u = idx & 63;
      ldsWe[u * LDS_STRIDE + f] = f2bf(wep[f * U_DIM + u]);
    }
    __syncthreads();

    v8f hacc[4];
#pragma unroll
    for (int nt = 0; nt < 4; ++nt) hacc[nt] = zero;
#pragma unroll
    for (int kt = 0; kt < 4; ++kt)
#pragma unroll
      for (int nt = 0; nt < 4; ++nt)
        hacc[nt] = wmma_bf16(a[kt], load_b_lds(ldsWe + (nt * 16 + n) * LDS_STRIDE, kt, h), hacc[nt]);

    float gl[8];
#pragma unroll
    for (int j = 0; j < 8; ++j) {
      float gv = (e < 16) ? g0[j] : g1[j];
      gl[j] = __shfl(gv, (lane & 16) | (e & 15), 32);
    }
#pragma unroll
    for (int nt = 0; nt < 4; ++nt) {
      float bev = be[e * U_DIM + nt * 16 + n];
#pragma unroll
      for (int j = 0; j < 8; ++j) {
        float hv = hacc[nt][j] + bev;
        hv = (hv >= 0.0f) ? hv : 0.01f * hv;
        oacc[nt][j] = __builtin_fmaf(hv, gl[j], oacc[nt][j]);
      }
    }
  }

  float* op = out + (size_t)tw * U_DIM;
#pragma unroll
  for (int nt = 0; nt < 4; ++nt)
#pragma unroll
    for (int j = 0; j < 8; ++j)
      op[(j + 8 * h) * U_DIM + nt * 16 + n] = oacc[nt][j];
}

// ---------------------------------------------------------------------------
extern "C" void kernel_launch(void* const* d_in, const int* in_sizes, int n_in,
                              void* d_out, int out_size, void* d_ws, size_t ws_size,
                              hipStream_t stream) {
  (void)in_sizes; (void)n_in; (void)out_size;
  const float* x  = (const float*)d_in[0];  // [32,512,128]
  const float* Wg = (const float*)d_in[1];  // [128,32]
  const float* bg = (const float*)d_in[2];  // [32]
  const float* We = (const float*)d_in[3];  // [32,128,64]
  const float* be = (const float*)d_in[4];  // [32,64]
  float* out = (float*)d_out;               // [32,512,64]

  if (ws_size >= WS_BYTES && d_ws != nullptr) {
    unsigned short* ws = (unsigned short*)d_ws;
    dim3 pgrid((WS_ELEMS + 255) / 256), pblock(256);
    moe_prep_kernel<<<pgrid, pblock, 0, stream>>>(x, Wg, We, ws);
    dim3 grid(TOKENS / 64), block(128);    // 256 WGs, 4 wave32 each
    moe_main_kernel<<<grid, block, 0, stream>>>(ws, bg, be, out);
  } else {
    dim3 grid(TOKENS / 128), block(256);
    moe_fallback_kernel<<<grid, block, 0, stream>>>(x, Wg, bg, We, be, out);
  }
}
